// BehaviorMoE_90374701843251
// MI455X (gfx1250) — matrix operand order repaired and checked by reading, not compiled, not run on hardware
//
#include <hip/hip_runtime.h>
#include <hip/hip_bf16.h>

#define D_MODEL  1024
#define N_TOK    8192
#define N_B      4
#define N_E_SH   3
#define LN_EPS   1e-5f

typedef __attribute__((ext_vector_type(16))) __bf16 v16bf;
typedef __attribute__((ext_vector_type(8)))  __bf16 v8bf;
typedef __attribute__((ext_vector_type(4)))  __bf16 v4bf;
typedef __attribute__((ext_vector_type(8)))  float  v8f;

// ---------------- workspace layout (bytes) ----------------
#define WS_GATES   0u
#define WS_COUNTS  131072u
#define WS_LISTS   131328u
#define WS_SEL     262400u                       // 32 MB
#define WS_XHI     (WS_SEL + 33554432u)          // 16 MB
#define WS_XLO     (WS_XHI + 16777216u)          // 16 MB
#define WS_WHI     (WS_XLO + 16777216u)          // 14 MB (7 x 1024 x 1024 bf16)
#define WS_WLO     (WS_WHI + 14680064u)          // 14 MB

__global__ __launch_bounds__(32)
void bmoe_zero_counts(int* __restrict__ counts) {
    if (threadIdx.x < 4) counts[threadIdx.x] = 0;
}

// ---- fp32 -> (bf16 hi, bf16 lo) split, 4 elements/thread ----
__global__ __launch_bounds__(256)
void bmoe_split(const float* __restrict__ src, __bf16* __restrict__ hi,
                __bf16* __restrict__ lo)
{
    const size_t i = ((size_t)blockIdx.x * 256 + threadIdx.x) * 4;
    float4 f = *(const float4*)(src + i);
    float v[4] = { f.x, f.y, f.z, f.w };
    v4bf h, l;
#pragma unroll
    for (int k = 0; k < 4; ++k) {
        __bf16 hh = (__bf16)v[k];
        h[k] = hh;
        l[k] = (__bf16)(v[k] - (float)hh);
    }
    *(v4bf*)(hi + i) = h;
    *(v4bf*)(lo + i) = l;
}

// ---- gating: one wave per token ----
__global__ __launch_bounds__(256)
void bmoe_gate(const float* __restrict__ x, const int* __restrict__ b_seq,
               const float* __restrict__ w_gates,
               float* __restrict__ gates, int* __restrict__ counts,
               int* __restrict__ lists)
{
    const int wave = threadIdx.x >> 5, lane = threadIdx.x & 31;
    const int n = blockIdx.x * 8 + wave;
    const int b = b_seq[n];
    if (b == 0) return;
    const int t = b - 1;

    const float* xr = x + (size_t)n * D_MODEL;
    const float* wg = w_gates + (size_t)t * D_MODEL * 4;
    float a0 = 0.f, a1 = 0.f, a2 = 0.f, a3 = 0.f;
    for (int d = lane; d < D_MODEL; d += 32) {
        float  xv = xr[d];
        float4 w  = *(const float4*)(wg + d * 4);
        a0 += xv * w.x; a1 += xv * w.y; a2 += xv * w.z; a3 += xv * w.w;
    }
#pragma unroll
    for (int off = 16; off; off >>= 1) {
        a0 += __shfl_xor(a0, off, 32);
        a1 += __shfl_xor(a1, off, 32);
        a2 += __shfl_xor(a2, off, 32);
        a3 += __shfl_xor(a3, off, 32);
    }
    if (lane == 0) {
        float m  = fmaxf(fmaxf(a0, a1), fmaxf(a2, a3));
        float e0 = expf(a0 - m), e1 = expf(a1 - m);
        float e2 = expf(a2 - m), e3 = expf(a3 - m);
        float inv = 1.f / (e0 + e1 + e2 + e3);
        *(float4*)(gates + (size_t)n * 4) =
            make_float4(e0 * inv, e1 * inv, e2 * inv, e3 * inv);
        int pos = atomicAdd(&counts[t], 1);
        lists[t * N_TOK + pos] = n;
    }
}

// A fragment: one 32B contiguous run
__device__ __forceinline__ v16bf bmoe_ldA(const __bf16* p) {
    v8bf r0 = *(const v8bf*)(p);
    v8bf r1 = *(const v8bf*)(p + 8);
    return __builtin_shufflevector(r0, r1, 0,1,2,3,4,5,6,7,8,9,10,11,12,13,14,15);
}
// B fragment: one 32B contiguous run
__device__ __forceinline__ v16bf bmoe_ldB(const __bf16* p) {
    v8bf r0 = *(const v8bf*)(p);
    v8bf r1 = *(const v8bf*)(p + 8);
    return __builtin_shufflevector(r0, r1, 0,1,2,3,4,5,6,7,8,9,10,11,12,13,14,15);
}

// ---- fused grouped expert GEMM + gated combine ----
// grid: (D_MODEL/128, N_TOK/64, N_B), block 256 (8 waves)
// per block: BM=64 gathered tokens of one behavior, BN=128 output cols
__global__ __launch_bounds__(256)
void bmoe_gemm(const __bf16* __restrict__ xhi, const __bf16* __restrict__ xlo,
               const __bf16* __restrict__ whi, const __bf16* __restrict__ wlo,
               const float* __restrict__ b_sh, const float* __restrict__ b_sp,
               const float* __restrict__ gates,
               const int* __restrict__ counts, const int* __restrict__ lists,
               float* __restrict__ sel)
{
    const int t   = blockIdx.z;
    const int cnt = counts[t];
    const int m0  = blockIdx.y * 64;
    if (m0 >= cnt) return;

    __shared__ int   toks[64];
    __shared__ float gl[64][4];

    const int tid = threadIdx.x;
    if (tid < 64) {
        int gi  = m0 + tid;
        int tok = lists[t * N_TOK + (gi < cnt ? gi : m0)];
        toks[tid] = tok;
        float4 g  = *(const float4*)(gates + (size_t)tok * 4);
        gl[tid][0] = g.x; gl[tid][1] = g.y; gl[tid][2] = g.z; gl[tid][3] = g.w;
    }
    __syncthreads();

    const int wave   = tid >> 5;
    const int lane   = tid & 31;
    const int laneN  = lane & 15;
    const int laneHi = lane >> 4;
    const int wm     = wave >> 1;           // 0..3 : 16-row strip
    const int wn     = wave & 1;            // 0..1 : 64-col strip
    const int nBase  = blockIdx.x * 128 + wn * 64;

    // A: 16x32 bf16 layout. lane<16: M=lane, K=0..15; lane>=16: K=16..31
    const size_t aOff = (size_t)toks[wm * 16 + laneN] * D_MODEL + laneHi * 16;
    const __bf16* aHiP = xhi + aOff;
    const __bf16* aLoP = xlo + aOff;

    v8f selAcc[4];
#pragma unroll
    for (int j = 0; j < 4; ++j)
#pragma unroll
        for (int r = 0; r < 8; ++r) selAcc[j][r] = 0.f;

#pragma unroll 1
    for (int e = 0; e < 4; ++e) {           // 3 shared + 1 behavior-specific
        size_t planeOff; const float* be; int gcol;
        if (e < N_E_SH) {
            planeOff = (size_t)e * D_MODEL * D_MODEL;
            be = b_sh + e * D_MODEL; gcol = e;
        } else {
            planeOff = (size_t)(N_E_SH + t) * D_MODEL * D_MODEL;
            be = b_sp + t * D_MODEL; gcol = 3;
        }

        v8f cur[4];
#pragma unroll
        for (int j = 0; j < 4; ++j)
#pragma unroll
            for (int r = 0; r < 8; ++r) cur[j][r] = 0.f;

        // B: 32x16 (KxN). lane<16: N=lane, K=0..15; lane>=16: K=16..31.
        // y = x @ W^T => B[k][n] = W[n][k] : contiguous along K (row-major W).
        const size_t bBase = planeOff + (size_t)laneHi * 16;

        for (int k0 = 0; k0 < D_MODEL; k0 += 32) {
            v16bf ahi = bmoe_ldA(aHiP + k0);
            v16bf alo = bmoe_ldA(aLoP + k0);

#pragma unroll
            for (int j = 0; j < 4; ++j) {
                const size_t bo = bBase + (size_t)(nBase + j * 16 + laneN) * D_MODEL + k0;
                v16bf bhi = bmoe_ldB(whi + bo);
                v16bf blo = bmoe_ldB(wlo + bo);

                cur[j] = __builtin_amdgcn_wmma_f32_16x16x32_bf16(
                             false, ahi, false, bhi, (short)0, cur[j], false, false);
                cur[j] = __builtin_amdgcn_wmma_f32_16x16x32_bf16(
                             false, ahi, false, blo, (short)0, cur[j], false, false);
                cur[j] = __builtin_amdgcn_wmma_f32_16x16x32_bf16(
                             false, alo, false, bhi, (short)0, cur[j], false, false);
            }
        }

        // gated accumulate: sel += g_e * (acc + bias_e)
#pragma unroll
        for (int j = 0; j < 4; ++j) {
            const float bias = be[nBase + j * 16 + laneN];
#pragma unroll
            for (int r = 0; r < 8; ++r) {
                const int M = wm * 16 + laneHi * 8 + r;   // C/D layout row
                selAcc[j][r] += gl[M][gcol] * (cur[j][r] + bias);
            }
        }
    }

    // scatter to sel[token, col]
#pragma unroll
    for (int j = 0; j < 4; ++j) {
        const int col = nBase + j * 16 + laneN;
#pragma unroll
        for (int r = 0; r < 8; ++r) {
            const int M = wm * 16 + laneHi * 8 + r;
            if (m0 + M < cnt)
                sel[(size_t)toks[M] * D_MODEL + col] = selAcc[j][r];
        }
    }
}

// ---- LayerNorm + residual: one wave per token ----
__global__ __launch_bounds__(256)
void bmoe_ln(const float* __restrict__ x, const int* __restrict__ b_seq,
             const float* __restrict__ sel, const float* __restrict__ gamma,
             const float* __restrict__ beta, float* __restrict__ out)
{
    const int wave = threadIdx.x >> 5, lane = threadIdx.x & 31;
    const int n = blockIdx.x * 8 + wave;
    const float* xr   = x   + (size_t)n * D_MODEL;
    float*       orow = out + (size_t)n * D_MODEL;

    if (b_seq[n] == 0) {                   // sel == 0 -> LN(0) = beta
        for (int d = lane; d < D_MODEL; d += 32)
            orow[d] = xr[d] + beta[d];
        return;
    }

    const float* sr = sel + (size_t)n * D_MODEL;
    float sv[32];
    float s = 0.f, ss = 0.f;
#pragma unroll
    for (int i = 0; i < 32; ++i) {
        float v = sr[lane + i * 32];
        sv[i] = v; s += v; ss += v * v;
    }
#pragma unroll
    for (int off = 16; off; off >>= 1) {
        s  += __shfl_xor(s,  off, 32);
        ss += __shfl_xor(ss, off, 32);
    }
    const float mu  = s  * (1.f / D_MODEL);
    const float var = ss * (1.f / D_MODEL) - mu * mu;
    const float rs  = rsqrtf(var + LN_EPS);
#pragma unroll
    for (int i = 0; i < 32; ++i) {
        const int d = lane + i * 32;
        orow[d] = xr[d] + (sv[i] - mu) * rs * gamma[d] + beta[d];
    }
}

extern "C" void kernel_launch(void* const* d_in, const int* in_sizes, int n_in,
                              void* d_out, int out_size, void* d_ws, size_t ws_size,
                              hipStream_t stream)
{
    const float* x       = (const float*)d_in[0];
    const int*   b_seq   = (const int*)  d_in[1];
    const float* W_sh    = (const float*)d_in[2];
    const float* b_sh    = (const float*)d_in[3];
    const float* W_sp    = (const float*)d_in[4];
    const float* b_sp    = (const float*)d_in[5];
    const float* w_gates = (const float*)d_in[6];
    const float* gamma   = (const float*)d_in[7];
    const float* beta    = (const float*)d_in[8];
    float* out = (float*)d_out;

    char*   ws     = (char*)d_ws;
    float*  gates  = (float*) (ws + WS_GATES);
    int*    counts = (int*)   (ws + WS_COUNTS);
    int*    lists  = (int*)   (ws + WS_LISTS);
    float*  sel    = (float*) (ws + WS_SEL);
    __bf16* xhi    = (__bf16*)(ws + WS_XHI);
    __bf16* xlo    = (__bf16*)(ws + WS_XLO);
    __bf16* whi    = (__bf16*)(ws + WS_WHI);
    __bf16* wlo    = (__bf16*)(ws + WS_WLO);

    bmoe_zero_counts<<<1, 32, 0, stream>>>(counts);

    // precompute bf16 hi/lo planes (done once; removes all VALU from GEMM loop)
    bmoe_split<<<(N_TOK * D_MODEL) / 1024, 256, 0, stream>>>(x, xhi, xlo);
    bmoe_split<<<(N_E_SH * D_MODEL * D_MODEL) / 1024, 256, 0, stream>>>(W_sh, whi, wlo);
    bmoe_split<<<(N_B * D_MODEL * D_MODEL) / 1024, 256, 0, stream>>>(
        W_sp, whi + (size_t)N_E_SH * D_MODEL * D_MODEL,
              wlo + (size_t)N_E_SH * D_MODEL * D_MODEL);

    bmoe_gate<<<N_TOK / 8, 256, 0, stream>>>(x, b_seq, w_gates, gates, counts, lists);

    dim3 g2(D_MODEL / 128, N_TOK / 64, N_B);
    bmoe_gemm<<<g2, 256, 0, stream>>>(xhi, xlo, whi, wlo, b_sh, b_sp,
                                      gates, counts, lists, sel);

    bmoe_ln<<<N_TOK / 8, 256, 0, stream>>>(x, b_seq, sel, gamma, beta, out);
}
